// OrthoLinear_24455543783708
// MI455X (gfx1250) — matrix operand order repaired and checked by reading, not compiled, not run on hardware
//
#include <hip/hip_runtime.h>
#include <hip/hip_fp16.h>
#include <hip/hip_bf16.h>

// Problem constants (from reference)
#define IN_F  4096
#define OUT_F 4096

// GEMM tiling
#define BM 128
#define BN 128
#define BK 64

typedef __attribute__((ext_vector_type(16))) __bf16        v16bf;
typedef __attribute__((ext_vector_type(8)))  float         v8f;
typedef __attribute__((ext_vector_type(4)))  int           v4i;
typedef __attribute__((ext_vector_type(4)))  unsigned int  v4u;
typedef __attribute__((ext_vector_type(8)))  unsigned int  v8u;
typedef __attribute__((ext_vector_type(4)))  unsigned short v4us;

// fp32 -> bf16 via native conversion (RNE; lowers to v_cvt_pk_bf16_f32-class ops)
static __device__ __forceinline__ unsigned short f2bf(float f) {
    __bf16 h = (__bf16)f;
    return __builtin_bit_cast(unsigned short, h);
}

static __device__ __forceinline__ v4us cvt4(float4 f) {
    v4us h;
    h[0] = f2bf(f.x); h[1] = f2bf(f.y);
    h[2] = f2bf(f.z); h[3] = f2bf(f.w);
    return h;
}

// Build a 16-element bf16 fragment from two 16-byte LDS chunks
static __device__ __forceinline__ v16bf load_frag(const unsigned short* p0,
                                                  const unsigned short* p1) {
    v4u lo = *(const v4u*)p0;
    v4u hi = *(const v4u*)p1;
    v8u cat;
#pragma unroll
    for (int i = 0; i < 4; ++i) { cat[i] = lo[i]; cat[i + 4] = hi[i]; }
    return __builtin_bit_cast(v16bf, cat);
}

// ---------------------------------------------------------------------------
// Kernel 1: W[row,:] = (u8 - 128) * scale[row]  (+ CSR residual scatter), bf16
// One workgroup per output row; ds_add_f32 atomics handle duplicate columns
// exactly like the reference's .at[].add.
// ---------------------------------------------------------------------------
__global__ __launch_bounds__(256)
void dequant_scatter_kernel(const int*   __restrict__ base,
                            const float* __restrict__ scales,
                            const __half* __restrict__ vals,
                            const int*   __restrict__ cols,
                            const int*   __restrict__ rowptr,
                            unsigned short* __restrict__ W) {
    __shared__ float wrow[IN_F];
    const int row = blockIdx.x;
    const float s = scales[row];
    const int* brow = base + (size_t)row * IN_F;

    for (int c = threadIdx.x * 4; c < IN_F; c += blockDim.x * 4) {
        int4 b = *(const int4*)(brow + c);
        float4 f;
        f.x = (float)(b.x - 128) * s;
        f.y = (float)(b.y - 128) * s;
        f.z = (float)(b.z - 128) * s;
        f.w = (float)(b.w - 128) * s;
        *(float4*)&wrow[c] = f;
    }
    __syncthreads();

    const int e0 = rowptr[row], e1 = rowptr[row + 1];
    for (int i = e0 + (int)threadIdx.x; i < e1; i += blockDim.x) {
        atomicAdd(&wrow[cols[i]], __half2float(vals[i]));   // ds_add_f32
    }
    __syncthreads();

    unsigned short* wout = W + (size_t)row * IN_F;
    for (int c = threadIdx.x * 4; c < IN_F; c += blockDim.x * 4) {
        *(v4us*)(wout + c) = cvt4(*(const float4*)&wrow[c]);
    }
}

// ---------------------------------------------------------------------------
// Kernel 2: out[T, OUT_F] = x @ W^T, bf16 WMMA, fp32 acc, double-buffered LDS.
// B panel staged with GLOBAL_LOAD_ASYNC_TO_LDS_B128 (ASYNCcnt), A panel
// converted fp32->bf16 in VALU (co-executes with XDL WMMA) and ds_store'd.
// One barrier per K-step.
// ---------------------------------------------------------------------------
__global__ __launch_bounds__(256)
void gemm_bf16_kernel(const float* __restrict__ X,
                      const unsigned short* __restrict__ W,
                      float* __restrict__ out) {
    __shared__ unsigned short As[2][BM][BK];   // 2 x 16 KB
    __shared__ unsigned short Bs[2][BN][BK];   // 2 x 16 KB

    const int tid   = threadIdx.x;
    const int lane  = tid & 31;
    const int wave  = tid >> 5;      // 0..7
    const int waveM = wave >> 2;     // 0..1  -> 64-row band
    const int waveN = wave & 3;      // 0..3  -> 32-col band
    const int hi    = lane >> 4;
    const int l15   = lane & 15;

    const int mBase = blockIdx.y * BM;
    const int nBase = blockIdx.x * BN;

    v8f acc[4][2] = {};

    // Global->LDS staging maps
    const int aCol4 = (tid & 15) << 2;   // fp32 k offset (16B aligned)
    const int aRow0 = tid >> 4;          // 0..15, 8 rows/thread stride 16
    const int bCol8 = (tid & 7) << 3;    // bf16 k offset (16B aligned)
    const int bRow0 = tid >> 3;          // 0..31, 4 rows/thread stride 32

    // Hoisted per-thread base pointers: inner-loop addresses become
    // base + it*BK + constant row offsets (fits the signed-24-bit IOFFSET).
    const float*          aBase = X + (size_t)(mBase + aRow0) * IN_F + aCol4;
    const unsigned short* bBase = W + (size_t)(nBase + bRow0) * IN_F + bCol8;

    auto stageB = [&](int k0, int buf) {
#pragma unroll
        for (int i = 0; i < 4; ++i) {
            __builtin_amdgcn_global_load_async_to_lds_b128(
                (v4i*)(bBase + k0 + (size_t)i * (32 * IN_F)),
                (v4i*)&Bs[buf][bRow0 + i * 32][bCol8], 0, 0);
        }
    };

    auto stageA = [&](int k0, int buf) {
#pragma unroll
        for (int i = 0; i < 8; ++i) {
            float4 f = *(const float4*)(aBase + k0 + (size_t)i * (16 * IN_F));
            *(v4us*)&As[buf][aRow0 + i * 16][aCol4] = cvt4(f);
        }
    };

    auto compute = [&](int buf) {
#pragma unroll
        for (int kk = 0; kk < BK; kk += 32) {
            // A frag 16x32: lanes 0-15 hold K[0..7],K[16..23]; 16-31 K[8..15],K[24..31]
            v16bf afr[4];
#pragma unroll
            for (int mi = 0; mi < 4; ++mi) {
                const int r = waveM * 64 + mi * 16 + l15;
                afr[mi] = load_frag(&As[buf][r][kk + hi * 8],
                                    &As[buf][r][kk + 16 + hi * 8]);
            }
            // B frag 32x16: col = lane&15; lanes 0-15 hold K[0..7],K[16..23]; 16-31 K[8..15],K[24..31]
            v16bf bfr[2];
#pragma unroll
            for (int ni = 0; ni < 2; ++ni) {
                const int c = waveN * 32 + ni * 16 + l15;
                bfr[ni] = load_frag(&Bs[buf][c][kk + hi * 8],
                                    &Bs[buf][c][kk + 16 + hi * 8]);
            }
#pragma unroll
            for (int mi = 0; mi < 4; ++mi)
#pragma unroll
                for (int ni = 0; ni < 2; ++ni)
                    acc[mi][ni] = __builtin_amdgcn_wmma_f32_16x16x32_bf16(
                        false, afr[mi], false, bfr[ni],
                        (short)0, acc[mi][ni], false, false);
        }
    };

    const int KITERS = IN_F / BK;

    // Prologue: fill buffer 0
    stageB(0, 0);
    stageA(0, 0);
    asm volatile("s_wait_asynccnt 0x0" ::: "memory");
    __syncthreads();

    for (int it = 0; it < KITERS; ++it) {
        const int cur = it & 1;
        if (it + 1 < KITERS) {
            stageB((it + 1) * BK, cur ^ 1);     // async, overlaps WMMA below
            stageA((it + 1) * BK, cur ^ 1);     // loads/cvts interleave with WMMA
            __builtin_prefetch(aBase + (it + 2) * BK, 0, 3);
        }
        compute(cur);
        asm volatile("s_wait_asynccnt 0x0" ::: "memory");
        __syncthreads();
    }

    // Store C: VGPR v -> M = v + 8*hi, N = lane&15 (ISA 32-bit C/D layout)
#pragma unroll
    for (int mi = 0; mi < 4; ++mi) {
#pragma unroll
        for (int ni = 0; ni < 2; ++ni) {
            const int col   = nBase + waveN * 32 + ni * 16 + l15;
            const int rbase = mBase + waveM * 64 + mi * 16 + hi * 8;
#pragma unroll
            for (int v = 0; v < 8; ++v)
                out[(size_t)(rbase + v) * OUT_F + col] = acc[mi][ni][v];
        }
    }
}

// ---------------------------------------------------------------------------
extern "C" void kernel_launch(void* const* d_in, const int* in_sizes, int n_in,
                              void* d_out, int out_size, void* d_ws, size_t ws_size,
                              hipStream_t stream) {
    const float*  x            = (const float*)d_in[0];
    const int*    base_packed  = (const int*)d_in[1];
    const float*  scales       = (const float*)d_in[2];
    const __half* ortho_vals   = (const __half*)d_in[3];
    const int*    ortho_idx    = (const int*)d_in[4];
    const int*    ortho_ptr    = (const int*)d_in[5];
    float*        out          = (float*)d_out;

    unsigned short* W = (unsigned short*)d_ws;   // bf16 W, 4096*4096*2 = 32 MiB

    dequant_scatter_kernel<<<OUT_F, 256, 0, stream>>>(
        base_packed, scales, ortho_vals, ortho_idx, ortho_ptr, W);

    const int T = in_sizes[0] / IN_F;            // 8192 tokens
    dim3 grid(OUT_F / BN, T / BM);
    gemm_bf16_kernel<<<grid, 256, 0, stream>>>(x, W, out);
}
